// MLP_VSA_Layer_55284819034113
// MI455X (gfx1250) — matrix operand order repaired and checked by reading, not compiled, not run on hardware
//
#include <hip/hip_runtime.h>

#define BEV_H 160
#define BEV_W 160
#define BEV_B 4
#define NPIX  (BEV_B * BEV_H * BEV_W)   // 102400
#define CCH   512

typedef __attribute__((ext_vector_type(16))) _Float16 v16h;
typedef __attribute__((ext_vector_type(8)))  _Float16 v8h;
typedef __attribute__((ext_vector_type(8)))  float    v8f;
typedef __attribute__((ext_vector_type(4)))  unsigned int u32x4;
typedef __attribute__((ext_vector_type(8)))  unsigned int u32x8;

// ---------------------------------------------------------------- WMMA utils
__device__ inline v8f wmma_f16(v16h a, v16h b, v8f c) {
  return __builtin_amdgcn_wmma_f32_16x16x32_f16(false, a, false, b, (short)0, c,
                                                false, false);
}

// A fragment: 16x32 f16, row-major source with row stride S halves.
template <int S>
__device__ inline v16h afrag(const _Float16* A, int lane, int kb) {
  const int m = lane & 15, g = (lane >> 4) & 1;
  const _Float16* p = A + m * S + kb + g * 8;
  union { v16h v; v8h h[2]; } u;
  u.h[0] = *(const v8h*)p;
  u.h[1] = *(const v8h*)(p + 16);
  return u.v;
}

// B fragment: 32x16 f16 from a transposed (N-major, K-contiguous) panel.
template <int S>
__device__ inline v16h bfrag(const _Float16* Wt, int lane, int ncol, int kb) {
  const int n = lane & 15, g = (lane >> 4) & 1;
  const _Float16* p = Wt + (ncol + n) * S + kb + g * 8;
  union { v16h v; v8h h[2]; } u;
  u.h[0] = *(const v8h*)p;
  u.h[1] = *(const v8h*)(p + 16);
  return u.v;
}

// One 16x16 output tile of A[16x64] @ W[64x64] (K=64 -> two WMMAs).
template <int SA, int SB>
__device__ inline v8f wmma_coltile(const _Float16* A, const _Float16* Wt,
                                   int lane, int n4) {
  v16h a0 = afrag<SA>(A, lane, 0);
  v16h a1 = afrag<SA>(A, lane, 32);
  v16h b0 = bfrag<SB>(Wt, lane, n4 * 16, 0);
  v16h b1 = bfrag<SB>(Wt, lane, n4 * 16, 32);
  v8f c = {};
  c = wmma_f16(a0, b0, c);
  c = wmma_f16(a1, b1, c);
  return c;
}

// ------------------------------------------- CDNA5 async / TDM data movement
// Async global->LDS (16B per op); tracked with ASYNCcnt.
__device__ inline void async_ld_b128(unsigned lds_byte_off, const void* gaddr) {
  asm volatile("global_load_async_to_lds_b128 %0, %1, off"
               :: "v"(lds_byte_off),
                  "v"((unsigned long long)(size_t)gaddr)
               : "memory");
}
__device__ inline void wait_async0() {
  asm volatile("s_wait_asynccnt 0x0" ::: "memory");
}

// TDM 2D tile load (D# groups per CDNA5 ISA 8.3/8.4). data_size = 2 bytes.
// Loads tile_d1 rows of tile_d0 elements (row stride stride_d0 elements),
// optionally inserting LDS padding every 2^(pad_iv+1) DWORDs.
__device__ inline void tdm_load_2d_f16(unsigned lds_byte_off, const void* gaddr,
                                       unsigned tile_d0, unsigned tile_d1,
                                       unsigned tensor_d0, unsigned tensor_d1,
                                       unsigned stride_d0,
                                       unsigned pad_en, unsigned pad_iv,
                                       unsigned pad_amt) {
  unsigned long long ga = (unsigned long long)(size_t)gaddr;
  u32x4 g0;
  g0[0] = 1u;                                   // count=1, user descriptor
  g0[1] = lds_byte_off;                         // lds_addr
  g0[2] = (unsigned)(ga & 0xFFFFFFFFu);         // global_addr[31:0]
  g0[3] = (unsigned)((ga >> 32) & 0x1FFFFFFu)   // global_addr[56:32]
          | (2u << 30);                         // type = 2 ("image")
  u32x8 g1;
  g1[0] = (1u << 16)                            // data_size = 1 -> 2 bytes
          | (pad_en << 20) | (pad_iv << 22) | (pad_amt << 25);
  g1[1] = (tensor_d0 & 0xFFFFu) << 16;          // tensor_dim0[15:0]
  g1[2] = ((tensor_d0 >> 16) & 0xFFFFu) | ((tensor_d1 & 0xFFFFu) << 16);
  g1[3] = ((tensor_d1 >> 16) & 0xFFFFu) | ((tile_d0 & 0xFFFFu) << 16);
  g1[4] = tile_d1 & 0xFFFFu;                    // tile_dim1 (tile_dim2 = 0)
  g1[5] = stride_d0;                            // tensor_dim0_stride[31:0]
  g1[6] = 0u;                                   // stride hi + dim1_stride lo
  g1[7] = 0u;
  asm volatile("tensor_load_to_lds %0, %1" :: "s"(g0), "s"(g1) : "memory");
}

__device__ inline void atomicMaxF(float* addr, float v) {
  if (v >= 0.f) atomicMax((int*)addr, __float_as_int(v));
  else          atomicMin((unsigned int*)addr, __float_as_uint(v));
}

// ---------------------------------------------------------------- small ops
__global__ void k_fill(float* __restrict__ p, float v, int n) {
  int i = blockIdx.x * 256 + threadIdx.x;
  if (i < n) p[i] = v;
}

// pw (1,1,C,C) [cin][cout] f32 -> transposed f16 panel pwh[n][k] (stride 512)
__global__ void prep_pwh(const float* __restrict__ pw, _Float16* __restrict__ pwh) {
  int idx = blockIdx.x * 256 + threadIdx.x;
  if (idx < CCH * CCH) {
    int nn = idx >> 9, k = idx & 511;
    pwh[idx] = (_Float16)pw[k * CCH + nn];
  }
}

// All 64x64 weight mats -> f16 transposed stride-72 panels in one buffer:
// [W1 W2 W3 Wq](4x4608h) [Ws 16x72](1152h) [Wk Wv Wo](3x4608h) = 33408 halves
__global__ void prep_wpan(const float* __restrict__ W1, const float* __restrict__ W2,
                          const float* __restrict__ W3, const float* __restrict__ Wq,
                          const float* __restrict__ Ws,
                          const float* __restrict__ Wk, const float* __restrict__ Wv,
                          const float* __restrict__ Wo,
                          _Float16* __restrict__ wpan) {
  int idx = blockIdx.x * 256 + threadIdx.x;
  if (idx < 7 * 4096) {
    int mat = idx >> 12, e = idx & 4095, k = e >> 6, n = e & 63;
    const float* src[7] = {W1, W2, W3, Wq, Wk, Wv, Wo};
    const unsigned base[7] = {0, 4608, 9216, 13824, 19584, 24192, 28800};
    wpan[base[mat] + n * 72 + k] = (_Float16)src[mat][k * 64 + n];
  } else if (idx < 7 * 4096 + 1024) {
    int e = idx - 7 * 4096, k = e >> 4, n = e & 15;
    wpan[18432 + n * 72 + k] = (n < 8) ? (_Float16)Ws[k * 8 + n] : (_Float16)0.f;
  }
}

// ------------------------------------------------- k1: fused MLP + s + q
// Flat LDS layout (halves): WT[4][64][72] @0, WTS[16][72] @18432, AT[8][16][72] @19584
__global__ void __launch_bounds__(256) k1_mlp(
    const float* __restrict__ inp, const _Float16* __restrict__ wpan,
    const float* __restrict__ b1, const float* __restrict__ g1,
    const float* __restrict__ bb1, const float* __restrict__ m1,
    const float* __restrict__ v1,
    const float* __restrict__ b2, const float* __restrict__ g2,
    const float* __restrict__ bb2, const float* __restrict__ m2,
    const float* __restrict__ v2,
    const float* __restrict__ b3, const float* __restrict__ g3,
    const float* __restrict__ bb3, const float* __restrict__ m3,
    const float* __restrict__ v3,
    const float* __restrict__ bs, const float* __restrict__ bq,
    float* __restrict__ xg, float* __restrict__ qg, float* __restrict__ sg,
    int N) {
  __shared__ _Float16 LB[28800];
  const int t = threadIdx.x, lane = t & 31, w = t >> 5;

  // async-copy all weight panels (39168 B = 2448 x b128)
  for (int chk = t; chk < 2448; chk += 256)
    async_ld_b128(chk * 16u, (const char*)wpan + (size_t)chk * 16);

  _Float16* At = &LB[19584 + w * 1152];
  {  // stage 16 input rows per wave, f32 -> f16
    int m = lane & 15, h = lane >> 4;
    int p = blockIdx.x * 128 + w * 16 + m;
    const float* src = inp + (size_t)p * 64 + h * 32;
    for (int u = 0; u < 8; ++u) {
      float4 f = (p < N) ? ((const float4*)src)[u] : make_float4(0.f, 0.f, 0.f, 0.f);
      _Float16* d = &At[m * 72 + h * 32 + u * 4];
      d[0] = (_Float16)f.x; d[1] = (_Float16)f.y;
      d[2] = (_Float16)f.z; d[3] = (_Float16)f.w;
    }
  }
  wait_async0();
  __syncthreads();

  const int prow0 = blockIdx.x * 128 + w * 16;
  const int n = lane & 15, g = (lane >> 4) & 1;
  const float* LBi[3] = {b1, b2, b3};
  const float* LGm[3] = {g1, g2, g3};
  const float* LBt[3] = {bb1, bb2, bb3};
  const float* LM[3]  = {m1, m2, m3};
  const float* LV[3]  = {v1, v2, v3};

  for (int L = 0; L < 3; ++L) {
    v8f c[4];
    for (int n4 = 0; n4 < 4; ++n4)
      c[n4] = wmma_coltile<72, 72>(At, &LB[L * 4608], lane, n4);
    for (int n4 = 0; n4 < 4; ++n4) {
      int ch = n4 * 16 + n;
      float sc = LGm[L][ch] * __frsqrt_rn(LV[L][ch] + 1e-3f);
      float sh = LBt[L][ch] - LM[L][ch] * sc;
      float bb = LBi[L][ch];
      for (int r = 0; r < 8; ++r) {
        float val = (c[n4][r] + bb) * sc + sh;
        if (L < 2) val = fmaxf(val, 0.f);
        int mr = r + 8 * g;
        At[mr * 72 + ch] = (_Float16)val;
        if (L == 2) {
          int p = prow0 + mr;
          if (p < N) xg[(size_t)p * 64 + ch] = val;
        }
      }
    }
  }
  {  // s = x @ Ws + bs  (8 valid cols)
    v8f c = wmma_coltile<72, 72>(At, &LB[18432], lane, 0);
    if (n < 8) {
      float bb = bs[n];
      for (int r = 0; r < 8; ++r) {
        int p = prow0 + r + 8 * g;
        if (p < N) sg[(size_t)p * 8 + n] = c[r] + bb;
      }
    }
  }
  {  // q = x @ Wq + bq
    v8f c[4];
    for (int n4 = 0; n4 < 4; ++n4)
      c[n4] = wmma_coltile<72, 72>(At, &LB[3 * 4608], lane, n4);
    for (int n4 = 0; n4 < 4; ++n4) {
      int ch = n4 * 16 + n;
      float bb = bq[ch];
      for (int r = 0; r < 8; ++r) {
        int p = prow0 + r + 8 * g;
        if (p < N) qg[(size_t)p * 64 + ch] = c[n4][r] + bb;
      }
    }
  }
}

// ----------------------------------------------- k2: scatter softmax pieces
__global__ void k2_max(const float* __restrict__ sg, const int* __restrict__ inv,
                       float* __restrict__ smax, int N) {
  int i = blockIdx.x * 256 + threadIdx.x;
  if (i < N) {
    int vox = inv[i];
    for (int k = 0; k < 8; ++k) atomicMaxF(&smax[vox * 8 + k], sg[(size_t)i * 8 + k]);
  }
}

__global__ void k2_exp(const float* __restrict__ sg, const float* __restrict__ smax,
                       const int* __restrict__ inv, float* __restrict__ eg,
                       float* __restrict__ den, int N) {
  int idx = blockIdx.x * 256 + threadIdx.x;
  if (idx < N * 8) {
    int i = idx >> 3, k = idx & 7;
    int vox = inv[i];
    float e = __expf(sg[idx] - smax[vox * 8 + k]);
    eg[idx] = e;
    atomicAdd(&den[vox * 8 + k], e);
  }
}

__global__ void __launch_bounds__(64) k2_scatter(const float* __restrict__ eg,
                                                 const float* __restrict__ den,
                                                 const int* __restrict__ inv,
                                                 const float* __restrict__ xg,
                                                 float* __restrict__ xacc, int N) {
  int i = blockIdx.x;
  if (i >= N) return;
  int d = threadIdx.x;
  int vox = inv[i];
  float xv = xg[(size_t)i * 64 + d];
  for (int k = 0; k < 8; ++k) {
    float a = eg[(size_t)i * 8 + k] / den[vox * 8 + k];
    atomicAdd(&xacc[(size_t)vox * CCH + k * 64 + d], a * xv);
  }
}

// ----------------------------------------------- k3: sparse -> dense (relu)
__global__ void k3_dense(const float* __restrict__ xacc, const int* __restrict__ coords,
                         float* __restrict__ dense, int M) {
  int idx = blockIdx.x * 256 + threadIdx.x;
  int m = idx >> 7;
  if (m >= M) return;
  int c4 = (idx & 127) * 4;
  int bz = coords[m * 3 + 0], y = coords[m * 3 + 1], x = coords[m * 3 + 2];
  size_t pix = ((size_t)bz * BEV_H + y) * BEV_W + x;
  float4 v = *(const float4*)(xacc + (size_t)m * CCH + c4);
  v.x = fmaxf(v.x, 0.f); v.y = fmaxf(v.y, 0.f);
  v.z = fmaxf(v.z, 0.f); v.w = fmaxf(v.w, 0.f);
  *(float4*)(dense + pix * CCH + c4) = v;
}

// ---------------------------------------- k4/k5: depthwise 3x3 + BN + ReLU
template <typename OUT>
__global__ void __launch_bounds__(128) k_dwconv(
    const float* __restrict__ in, const float* __restrict__ wgt,
    const float* __restrict__ gg, const float* __restrict__ gb,
    const float* __restrict__ gm, const float* __restrict__ gv,
    OUT* __restrict__ out) {
  int pix = blockIdx.x;
  int bz  = pix / (BEV_H * BEV_W);
  int rem = pix - bz * (BEV_H * BEV_W);
  int y = rem / BEV_W, x = rem - y * BEV_W;
  int c = threadIdx.x * 4;
  float a0 = 0.f, a1 = 0.f, a2 = 0.f, a3 = 0.f;
  for (int ky = -1; ky <= 1; ++ky) {
    int yy = y + ky;
    if ((unsigned)yy >= BEV_H) continue;
    for (int kx = -1; kx <= 1; ++kx) {
      int xx = x + kx;
      if ((unsigned)xx >= BEV_W) continue;
      const float* ip = in + (((size_t)bz * BEV_H + yy) * BEV_W + xx) * CCH + c;
      const float* wp = wgt + ((ky + 1) * 3 + (kx + 1)) * CCH + c;
      float4 iv = *(const float4*)ip;
      float4 wv = *(const float4*)wp;
      a0 += iv.x * wv.x; a1 += iv.y * wv.y;
      a2 += iv.z * wv.z; a3 += iv.w * wv.w;
    }
  }
  __builtin_prefetch(in + (size_t)((pix + 1) % NPIX) * CCH + c, 0, 1);
  float acc[4] = {a0, a1, a2, a3};
  for (int u = 0; u < 4; ++u) {
    int ch = c + u;
    float sc = gg[ch] * __frsqrt_rn(gv[ch] + 1e-5f);
    float val = fmaxf((acc[u] - gm[ch]) * sc + gb[ch], 0.f);
    out[(size_t)pix * CCH + ch] = (OUT)val;
  }
}

// ----------------------------------------------- k6: pointwise 512x512 GEMM
// B panel (64 cols x 256 K-halves, +8 halves row pad) staged by the TDM.
__global__ void __launch_bounds__(256) k6_pw(const _Float16* __restrict__ Ain,
                                             const _Float16* __restrict__ Bw,
                                             float* __restrict__ Cout) {
  __shared__ _Float16 Bl[64 * 264];
  const int t = threadIdx.x, lane = t & 31, w = t >> 5;
  const int rb = blockIdx.x, nb = blockIdx.y;
  const int row0 = rb * 128 + w * 16;
  v8f acc[4] = {};
  for (int kh = 0; kh < 2; ++kh) {
    __syncthreads();
    if (w == 0) {
      // 64 rows x 256 halves from Bw[nb*64 .. +64)[kh*256 .. +256),
      // pad 4 DWORDs (8 halves) after every 128 DWORDs (one 256-half row).
      tdm_load_2d_f16(/*lds*/0u,
                      Bw + (size_t)(nb * 64) * CCH + kh * 256,
                      /*tile_d0*/256u, /*tile_d1*/64u,
                      /*tensor_d0*/CCH, /*tensor_d1*/CCH,
                      /*stride_d0*/CCH,
                      /*pad_en*/1u, /*pad_iv*/6u, /*pad_amt*/3u);
      __builtin_amdgcn_s_wait_tensorcnt(0);
    }
    __syncthreads();
    const int m = lane & 15, g = (lane >> 4) & 1;
    for (int ks = 0; ks < 8; ++ks) {
      const _Float16* ap =
          Ain + (size_t)(row0 + m) * CCH + kh * 256 + ks * 32 + g * 8;
      union { v16h v; v8h h[2]; } ua;
      ua.h[0] = *(const v8h*)ap;
      ua.h[1] = *(const v8h*)(ap + 16);
      for (int n4 = 0; n4 < 4; ++n4) {
        v16h b = bfrag<264>(&Bl[0], lane, n4 * 16, ks * 32);
        acc[n4] = wmma_f16(ua.v, b, acc[n4]);
      }
    }
  }
  const int n = lane & 15, g = (lane >> 4) & 1;
  for (int n4 = 0; n4 < 4; ++n4)
    for (int r = 0; r < 8; ++r)
      Cout[(size_t)(row0 + r + 8 * g) * CCH + nb * 64 + n4 * 16 + n] = acc[n4][r];
}

// -------------------------- k7: gather + bn_out + K/V + attention + O + concat
// Flat LDS (halves): HS@0 KK@4608 VV@9216 Wk@13824 Wv@18432 Wo@23040
//                    OT@27648 f32-scratch@28800 (LG 64 floats, ATW 64 floats)
__global__ void __launch_bounds__(256) k7_attn(
    const float* __restrict__ inp, const float* __restrict__ qg,
    const float* __restrict__ pwout, const int* __restrict__ inverse,
    const int* __restrict__ coords, const _Float16* __restrict__ wpan,
    const float* __restrict__ og, const float* __restrict__ ob,
    const float* __restrict__ om, const float* __restrict__ ov,
    const float* __restrict__ bk, const float* __restrict__ bv,
    const float* __restrict__ bo,
    float* __restrict__ out, int N) {
  __shared__ _Float16 LB[29056];
  float* LGs = (float*)&LB[28800];
  float* ATW = LGs + 64;
  const int t = threadIdx.x, lane = t & 31, w = t >> 5;
  const int ibase = blockIdx.x * 8;

  // async-copy Wk/Wv/Wo panels (27648 B = 1728 x b128) to LDS @13824 halves
  for (int chk = t; chk < 1728; chk += 256)
    async_ld_b128(27648u + chk * 16u,
                  (const char*)wpan + 39168 + (size_t)chk * 16);

  for (int idx = t; idx < 64 * 64; idx += 256) {  // gather + bn_out -> hs (f16)
    int rr = idx >> 6, d = idx & 63;
    int p = rr >> 3, kk2 = rr & 7;
    int i = ibase + p;
    float val = 0.f;
    if (i < N) {
      int vox = inverse[i];
      int bz = coords[vox * 3 + 0], yy = coords[vox * 3 + 1], xx = coords[vox * 3 + 2];
      size_t pix = ((size_t)bz * BEV_H + yy) * BEV_W + xx;
      float h = pwout[pix * CCH + kk2 * 64 + d];
      float sc = og[d] * __frsqrt_rn(ov[d] + 1e-3f);
      val = (h - om[d]) * sc + ob[d];
    }
    LB[rr * 72 + d] = (_Float16)val;
  }
  wait_async0();
  __syncthreads();

  {  // K and V GEMMs: 32 coltile jobs over 8 waves
    const int n = lane & 15, g = (lane >> 4) & 1;
    for (int jj = 0; jj < 4; ++jj) {
      int j = w * 4 + jj;
      int mat = j >> 4, rt = (j >> 2) & 3, n4 = j & 3;
      v8f c = wmma_coltile<72, 72>(&LB[rt * 16 * 72],
                                   &LB[13824 + mat * 4608], lane, n4);
      int ch = n4 * 16 + n;
      float bb = (mat ? bv : bk)[ch];
      _Float16* dst = &LB[(mat ? 9216 : 4608)];
      for (int r = 0; r < 8; ++r)
        dst[(rt * 16 + r + 8 * g) * 72 + ch] = (_Float16)(c[r] + bb);
    }
  }
  __syncthreads();

  if (t < 64) {  // logits = q . k / 8
    int p = t >> 3, kk2 = t & 7;
    int i = ibase + p;
    float acc = 0.f;
    if (i < N) {
      const float* qrow = qg + (size_t)i * 64;
      for (int d = 0; d < 64; ++d) acc += qrow[d] * (float)LB[4608 + (p * 8 + kk2) * 72 + d];
    }
    LGs[p * 8 + kk2] = acc * 0.125f;
  }
  __syncthreads();
  if (t < 64) {  // softmax over 8 slots
    int p = t >> 3, kk2 = t & 7;
    float mx = -1e30f;
    for (int u = 0; u < 8; ++u) mx = fmaxf(mx, LGs[p * 8 + u]);
    float sum = 0.f;
    for (int u = 0; u < 8; ++u) sum += __expf(LGs[p * 8 + u] - mx);
    ATW[p * 8 + kk2] = __expf(LGs[p * 8 + kk2] - mx) / sum;
  }
  __syncthreads();
  {  // o = attn * v -> OT tile (rows 8..15 zero-padded)
    int p = t >> 5, d0 = (t & 31) * 2;
    for (int u = 0; u < 2; ++u) {
      int d = d0 + u;
      float o = 0.f;
      for (int kk2 = 0; kk2 < 8; ++kk2)
        o += ATW[p * 8 + kk2] * (float)LB[9216 + (p * 8 + kk2) * 72 + d];
      LB[27648 + p * 72 + d] = (_Float16)o;
    }
    if (t < 128) {
      int rr = 8 + (t >> 4), d = (t & 15) * 4;
      for (int u = 0; u < 4; ++u) LB[27648 + rr * 72 + d + u] = (_Float16)0.f;
    }
  }
  __syncthreads();
  if (w < 4) {  // O projection: o @ Wo + bo
    const int n = lane & 15, g = (lane >> 4) & 1;
    v8f c = wmma_coltile<72, 72>(&LB[27648], &LB[23040], lane, w);
    int ch = w * 16 + n;
    float bb = bo[ch];
    for (int r = 0; r < 8; ++r) {
      int mr = r + 8 * g;
      if (mr < 8) {
        int i = ibase + mr;
        if (i < N) out[(size_t)i * 128 + 64 + ch] = c[r] + bb;
      }
    }
  }
  for (int idx = t; idx < 8 * 64; idx += 256) {  // concat input half
    int p = idx >> 6, d = idx & 63;
    int i = ibase + p;
    if (i < N) out[(size_t)i * 128 + d] = inp[(size_t)i * 64 + d];
  }
}

// ---------------------------------------------------------------- launcher
extern "C" void kernel_launch(void* const* d_in, const int* in_sizes, int n_in,
                              void* d_out, int out_size, void* d_ws, size_t ws_size,
                              hipStream_t stream) {
  const float* inp = (const float*)d_in[0];
  const float* W1 = (const float*)d_in[1];  const float* b1 = (const float*)d_in[2];
  const float* g1 = (const float*)d_in[3];  const float* bb1 = (const float*)d_in[4];
  const float* m1 = (const float*)d_in[5];  const float* v1 = (const float*)d_in[6];
  const float* W2 = (const float*)d_in[7];  const float* b2 = (const float*)d_in[8];
  const float* g2 = (const float*)d_in[9];  const float* bb2 = (const float*)d_in[10];
  const float* m2 = (const float*)d_in[11]; const float* v2 = (const float*)d_in[12];
  const float* W3 = (const float*)d_in[13]; const float* b3 = (const float*)d_in[14];
  const float* g3 = (const float*)d_in[15]; const float* bb3 = (const float*)d_in[16];
  const float* m3 = (const float*)d_in[17]; const float* v3 = (const float*)d_in[18];
  const float* Ws = (const float*)d_in[19]; const float* bs = (const float*)d_in[20];
  const float* dw1 = (const float*)d_in[21];
  const float* c1g = (const float*)d_in[22]; const float* c1b = (const float*)d_in[23];
  const float* c1m = (const float*)d_in[24]; const float* c1v = (const float*)d_in[25];
  const float* dw2 = (const float*)d_in[26];
  const float* c2g = (const float*)d_in[27]; const float* c2b = (const float*)d_in[28];
  const float* c2m = (const float*)d_in[29]; const float* c2v = (const float*)d_in[30];
  const float* pw = (const float*)d_in[31];
  const float* og = (const float*)d_in[32]; const float* ob = (const float*)d_in[33];
  const float* om = (const float*)d_in[34]; const float* ov = (const float*)d_in[35];
  const float* Wq = (const float*)d_in[36]; const float* bq = (const float*)d_in[37];
  const float* Wk = (const float*)d_in[38]; const float* bk = (const float*)d_in[39];
  const float* Wv = (const float*)d_in[40]; const float* bv = (const float*)d_in[41];
  const float* Wo = (const float*)d_in[42]; const float* bo = (const float*)d_in[43];
  const int* inverse = (const int*)d_in[44];
  const int* coords  = (const int*)d_in[45];
  const int N = in_sizes[0] / 64;
  const int M = in_sizes[45] / 3;
  float* out = (float*)d_out;

  char* ws = (char*)d_ws;
  size_t off = 0;
  auto take = [&](size_t bytes) -> char* {
    char* p = ws + off;
    off = (off + bytes + 255) & ~(size_t)255;
    return p;
  };
  _Float16* pwh  = (_Float16*)take((size_t)CCH * CCH * 2);
  _Float16* wpan = (_Float16*)take((size_t)33408 * 2);
  float* xg      = (float*)take((size_t)N * 64 * 4);
  float* qg      = (float*)take((size_t)N * 64 * 4);
  float* sg      = (float*)take((size_t)N * 8 * 4);
  float* eg      = (float*)take((size_t)N * 8 * 4);
  float* smax    = (float*)take((size_t)M * 8 * 4);
  float* den     = (float*)take((size_t)M * 8 * 4);
  float* denseA  = (float*)take((size_t)NPIX * CCH * 4);
  float* denseB  = (float*)take((size_t)NPIX * CCH * 4);
  _Float16* h2h  = (_Float16*)take((size_t)NPIX * CCH * 2);
  float* xacc = denseB;  // [M,512] accumulator aliases denseB
  (void)ws_size; (void)n_in; (void)out_size;

  hipMemsetAsync(denseA, 0, (size_t)NPIX * CCH * 4, stream);
  hipMemsetAsync(xacc, 0, (size_t)M * CCH * 4, stream);
  hipMemsetAsync(den, 0, (size_t)M * 8 * 4, stream);
  k_fill<<<(M * 8 + 255) / 256, 256, 0, stream>>>(smax, -1e30f, M * 8);
  prep_pwh<<<(CCH * CCH + 255) / 256, 256, 0, stream>>>(pw, pwh);
  prep_wpan<<<(7 * 4096 + 1024 + 255) / 256, 256, 0, stream>>>(
      W1, W2, W3, Wq, Ws, Wk, Wv, Wo, wpan);

  k1_mlp<<<(N + 127) / 128, 256, 0, stream>>>(
      inp, wpan, b1, g1, bb1, m1, v1, b2, g2, bb2, m2, v2,
      b3, g3, bb3, m3, v3, bs, bq, xg, qg, sg, N);

  k2_max<<<(N + 255) / 256, 256, 0, stream>>>(sg, inverse, smax, N);
  k2_exp<<<(N * 8 + 255) / 256, 256, 0, stream>>>(sg, smax, inverse, eg, den, N);
  k2_scatter<<<N, 64, 0, stream>>>(eg, den, inverse, xg, xacc, N);
  k3_dense<<<(M * 128 + 255) / 256, 256, 0, stream>>>(xacc, coords, denseA, M);

  k_dwconv<float><<<NPIX, 128, 0, stream>>>(denseA, dw1, c1g, c1b, c1m, c1v, denseB);
  k_dwconv<_Float16><<<NPIX, 128, 0, stream>>>(denseB, dw2, c2g, c2b, c2m, c2v, h2h);
  k6_pw<<<dim3(NPIX / 128, 8), 256, 0, stream>>>(h2h, pwh, denseA);

  k7_attn<<<(N + 7) / 8, 256, 0, stream>>>(inp, qg, denseA, inverse, coords, wpan,
                                           og, ob, om, ov, bk, bv, bo, out, N);
}
